// Model_45432164057760
// MI455X (gfx1250) — matrix operand order repaired and checked by reading, not compiled, not run on hardware
//
#include <hip/hip_runtime.h>
#include <hip/hip_bf16.h>

// ---------------------------------------------------------------------------
// Fused AR-flow kernel for MI455X (gfx1250, wave32, WMMA).
//   pre[n,l,h] = lagsx[n,:] . W1[l,h,:128] + xt[n,l]*W1[l,h,128] + b1[l,h]
//   res[n,l]   = sum_h leaky(pre)*W2[l,h] + b2[l]
//   logdet[n]  = sum_l log|sum_h W2*dact*w1_t|
// GEMM as bf16x3 split (hi*hi + hi*lo + lo*hi) on v_wmma_f32_16x16x32_bf16.
// W1 slabs double-buffered in LDS, prefetched with GLOBAL_LOAD_ASYNC_TO_LDS
// (ASYNCcnt) so DMA of slab l+1 overlaps the 48 WMMAs of slab l.
// ---------------------------------------------------------------------------

#define NB      32
#define LEN     512
#define TROWS   514      // LEN + LAGS
#define LATENT  64
#define HID     64
#define IN_DIM  129
#define KDIM    128      // LAGS*LATENT
#define SLOPE   0.2f

#define NTILE    64      // time rows per workgroup
#define NTHREADS 128     // 4 wave32
#define NWGS     (NB * (LEN / NTILE))   // 256

typedef __attribute__((ext_vector_type(16))) __bf16 v16bf;
typedef __attribute__((ext_vector_type(8)))  __bf16 v8bf;
typedef __attribute__((ext_vector_type(8)))  float  v8f;
typedef __attribute__((ext_vector_type(4)))  int    v4i;

// pointer-to-v4i in explicit address spaces (global = AS1, LDS = AS3)
typedef __attribute__((address_space(1))) v4i* gv4i_p;
typedef __attribute__((address_space(3))) v4i* lv4i_p;

// ----- per-l W1 slab image: hi plane then lo plane, rows padded to 272B ----
#define W1PITCH_BF 136                 // bf16 elems per h-row (128 data + 8 pad)
#define W1ROW_B    272                 // bytes per h-row
#define PLANE_B    (HID * W1ROW_B)     // 17408
#define LBLK_B     (2 * PLANE_B)       // 34816 per l (hi + lo)
#define LBLK_CHUNKS (LBLK_B / 16)      // 2176 x b128

// ----- LDS layout (dynamic shared), all offsets 16B aligned -----
#define XT_OFF    0                    // 64*65*4  = 16640 (pitch 65: conflict-free)
#define RES_OFF   16640                // 16640
#define BUF_OFF   33280                // 2 x 34816 (double-buffered W1 slabs)
#define W1T_OFF   102912               // 2*64*4
#define B1_OFF    103424               // 2*64*4
#define W2_OFF    103936               // 2*64*4
#define SMEM_BYTES 104448

// ----- CDNA5 async global->LDS DMA (guarded; falls back to vector copy) ----
#if defined(__has_builtin)
#if __has_builtin(__builtin_amdgcn_global_load_async_to_lds_b128)
#define HAS_ASYNC_LDS 1
#endif
#endif

static __device__ __forceinline__ void wait_async0() {
#if defined(__has_builtin) && __has_builtin(__builtin_amdgcn_s_wait_asynccnt)
    __builtin_amdgcn_s_wait_asynccnt(0);
#else
    asm volatile("s_wait_asynccnt 0x0" ::: "memory");
#endif
}

// round-to-nearest-even fp32 -> bf16 bits (storage only, no bf16 arithmetic)
static __device__ __forceinline__ unsigned short f2bf_bits(float f) {
    unsigned u = __builtin_bit_cast(unsigned, f);
    return (unsigned short)((u + 0x7FFFu + ((u >> 16) & 1u)) >> 16);
}
static __device__ __forceinline__ __bf16 f2bf(float f) {
    unsigned short s = f2bf_bits(f);
    return __builtin_bit_cast(__bf16, s);
}
static __device__ __forceinline__ float bf2f(__bf16 h) {
    unsigned short s = __builtin_bit_cast(unsigned short, h);
    unsigned u = ((unsigned)s) << 16;
    return __builtin_bit_cast(float, u);
}

// ---------------------------------------------------------------------------
// Pre-pass: emit W1[:, :, :128] as the exact LDS slab image:
//   per l: [hi: 64 rows x 272B][lo: 64 rows x 272B], pads zeroed.
// ---------------------------------------------------------------------------
#define SPLIT_ELEMS (LATENT * 2 * HID * W1PITCH_BF)   // 1,114,112 bf16 slots

__global__ __launch_bounds__(256) void w1_split_kernel(
    const float* __restrict__ W1, unsigned short* __restrict__ out)
{
    int idx = blockIdx.x * 256 + threadIdx.x;
    if (idx >= SPLIT_ELEMS) return;
    int e  = idx % W1PITCH_BF;
    int t  = idx / W1PITCH_BF;      // ((l*2 + plane)*64 + h)
    int h  = t & 63;
    int pl = (t >> 6) & 1;
    int l  = t >> 7;
    unsigned short v = 0;
    if (e < KDIM) {
        float w = W1[(l * HID + h) * IN_DIM + e];
        unsigned short hib = f2bf_bits(w);
        if (pl == 0) {
            v = hib;
        } else {
            float hif = __builtin_bit_cast(float, ((unsigned)hib) << 16);
            v = f2bf_bits(w - hif);
        }
    }
    out[idx] = v;
}

// ---------------------------------------------------------------------------
// Main fused kernel. One WG = 64 time rows of one batch, all 64 latents.
// ---------------------------------------------------------------------------
__global__ __launch_bounds__(NTHREADS) void ar_flow_kernel(
    const float* __restrict__ x,  const float* __restrict__ W1,
    const float* __restrict__ b1, const float* __restrict__ W2,
    const float* __restrict__ b2,
    float* __restrict__ res_out, float* __restrict__ log_out,
    const char* __restrict__ w1pre)   // padded slab image in d_ws, or nullptr
{
    extern __shared__ char smem[];
    float* xt_lds  = (float*)(smem + XT_OFF);
    float* res_lds = (float*)(smem + RES_OFF);
    float* w1t_s   = (float*)(smem + W1T_OFF);
    float* b1_s    = (float*)(smem + B1_OFF);
    float* w2_s    = (float*)(smem + W2_OFF);

    const int tid  = threadIdx.x;
    const int lane = tid & 31;
    const int wid  = tid >> 5;          // wave 0..3, owns rows [16*wid, 16*wid+16)
    const int q    = lane & 15;
    const int ksel = lane >> 4;

    const int bb = blockIdx.x >> 3;           // batch
    const int t0 = (blockIdx.x & 7) * NTILE;  // time-block base
    const int n0 = bb * LEN + t0;             // flat output row base

    const bool usePre = (w1pre != nullptr);

    // ---- stage xt slab: xt_lds[m][l] = x[b, t0+m+2, l] (fp32, pitch 65) ----
    for (int idx = tid; idx < NTILE * LATENT; idx += NTHREADS) {
        int m = idx >> 6, c = idx & 63;
        xt_lds[m * 65 + c] = x[(bb * TROWS + t0 + m + 2) * 64 + c];
    }

    // ---- build A fragments in registers, ISA 16-bit A 16x32 layout --------
    v16bf ahi[4], alo[4];
    {
        const int row = t0 + wid * 16 + q;
        #pragma unroll
        for (int kt = 0; kt < 4; ++kt) {
            #pragma unroll
            for (int half = 0; half < 2; ++half) {
                int Kb = kt * 32 + half * 16 + ksel * 8;
                int j = Kb >> 6, c0 = Kb & 63;
                const float* p = x + (bb * TROWS + row + j) * 64 + c0;
                float4 f0 = *(const float4*)p;
                float4 f1 = *(const float4*)(p + 4);
                float f[8] = {f0.x, f0.y, f0.z, f0.w, f1.x, f1.y, f1.z, f1.w};
                #pragma unroll
                for (int i = 0; i < 8; ++i) {
                    __bf16 hi = f2bf(f[i]);
                    ahi[kt][half * 8 + i] = hi;
                    alo[kt][half * 8 + i] = f2bf(f[i] - bf2f(hi));
                }
            }
        }
    }

    // ---- slab staging helpers (lambdas keep the pipeline readable) --------
    auto stage_w1 = [&](int ls, int bufIdx) {
        char* bufBase = smem + BUF_OFF + bufIdx * LBLK_B;
        if (usePre) {
            const char* g = w1pre + (size_t)ls * LBLK_B;
#if defined(HAS_ASYNC_LDS)
            for (int i = tid; i < LBLK_CHUNKS; i += NTHREADS) {
                __builtin_amdgcn_global_load_async_to_lds_b128(
                    (gv4i_p)(g + i * 16), (lv4i_p)(bufBase + i * 16), 0, 0);
            }
#else
            for (int i = tid; i < LBLK_CHUNKS; i += NTHREADS) {
                *(uint4*)(bufBase + i * 16) = *(const uint4*)(g + i * 16);
            }
#endif
        } else {
            __bf16* hi = (__bf16*)bufBase;
            __bf16* lo = (__bf16*)(bufBase + PLANE_B);
            for (int idx = tid; idx < HID * KDIM; idx += NTHREADS) {
                int h = idx >> 7, k = idx & 127;
                float w = W1[(ls * HID + h) * IN_DIM + k];
                __bf16 hb = f2bf(w);
                hi[h * W1PITCH_BF + k] = hb;
                lo[h * W1PITCH_BF + k] = f2bf(w - bf2f(hb));
            }
        }
    };
    auto stage_small = [&](int ls, int bufIdx) {
        if (tid < 64) {
            w1t_s[bufIdx * 64 + tid] = W1[(ls * HID + tid) * IN_DIM + KDIM];
            b1_s[bufIdx * 64 + tid]  = b1[ls * HID + tid];
            w2_s[bufIdx * 64 + tid]  = W2[ls * HID + tid];
        }
    };

    // ---- prologue: stage slab 0 into buffer 0 ----
    stage_w1(0, 0);
    stage_small(0, 0);
#if defined(HAS_ASYNC_LDS)
    if (usePre) wait_async0();
#endif
    __syncthreads();   // also orders the xt stage

    float logacc = 0.f;

    for (int l = 0; l < LATENT; ++l) {
        const int cur = l & 1;
        const int nxt = cur ^ 1;

        // ---- kick off DMA of slab l+1 while we compute slab l ----
        if (l + 1 < LATENT) {
            stage_w1(l + 1, nxt);
            stage_small(l + 1, nxt);
        }

        // ---- GEMM: 16 rows x 64 h per wave, K=128, bf16x3 ----
        const char* hiBase = smem + BUF_OFF + cur * LBLK_B;
        const char* loBase = hiBase + PLANE_B;
        v8f acc[4] = {};
        #pragma unroll
        for (int kt = 0; kt < 4; ++kt) {
            #pragma unroll
            for (int ht = 0; ht < 4; ++ht) {
                // B frag: lane col N = q, K at kt*32 + 8*ksel + {0..7} and +16 + {0..7}
                const char* ph = hiBase + (ht * 16 + q) * W1ROW_B + kt * 64 + ksel * 16;
                const char* pl = loBase + (ht * 16 + q) * W1ROW_B + kt * 64 + ksel * 16;
                v8bf bh0 = *(const v8bf*)ph;
                v8bf bh1 = *(const v8bf*)(ph + 32);
                v8bf bl0 = *(const v8bf*)pl;
                v8bf bl1 = *(const v8bf*)(pl + 32);
                v16bf bh = __builtin_shufflevector(bh0, bh1, 0,1,2,3,4,5,6,7,8,9,10,11,12,13,14,15);
                v16bf bl = __builtin_shufflevector(bl0, bl1, 0,1,2,3,4,5,6,7,8,9,10,11,12,13,14,15);
                acc[ht] = __builtin_amdgcn_wmma_f32_16x16x32_bf16(
                    false, ahi[kt], false, bh, (short)0, acc[ht], false, false);
                acc[ht] = __builtin_amdgcn_wmma_f32_16x16x32_bf16(
                    false, alo[kt], false, bh, (short)0, acc[ht], false, false);
                acc[ht] = __builtin_amdgcn_wmma_f32_16x16x32_bf16(
                    false, ahi[kt], false, bl, (short)0, acc[ht], false, false);
            }
        }

        // ---- epilogue: rank-1 + bias + leaky, reduce over h (lanes) ----
        float w2v[4], w1tv[4], b1v[4], g[4];
        #pragma unroll
        for (int ht = 0; ht < 4; ++ht) {
            int h = ht * 16 + q;
            w2v[ht]  = w2_s[cur * 64 + h];
            w1tv[ht] = w1t_s[cur * 64 + h];
            b1v[ht]  = b1_s[cur * 64 + h];
            g[ht]    = w2v[ht] * w1tv[ht];
        }
        const float b2v = b2[l];

        float myres = 0.f, myj = 1.f;
        #pragma unroll
        for (int r = 0; r < 8; ++r) {
            // C layout: elem r -> row m = r + 8*ksel, col h = ht*16 + q
            float xv = xt_lds[(wid * 16 + r + 8 * ksel) * 65 + l];
            float rsum = 0.f, jsum = 0.f;
            #pragma unroll
            for (int ht = 0; ht < 4; ++ht) {
                float pre = acc[ht][r] + xv * w1tv[ht] + b1v[ht];
                bool pos = (pre >= 0.f);
                rsum += (pos ? pre : SLOPE * pre) * w2v[ht];
                jsum += pos ? g[ht] : SLOPE * g[ht];
            }
            #pragma unroll
            for (int mm = 1; mm <= 8; mm <<= 1) {
                rsum += __shfl_xor(rsum, mm, 32);
                jsum += __shfl_xor(jsum, mm, 32);
            }
            if (q == r) { myres = rsum; myj = jsum; }
        }
        if (q < 8) {   // lane owns row m = 16*wid + q + 8*ksel
            int m = wid * 16 + q + 8 * ksel;
            res_lds[m * 65 + l] = myres + b2v;
            logacc += __logf(fabsf(myj));
        }

        // ---- close the pipeline stage: DMA done, then all waves synced ----
#if defined(HAS_ASYNC_LDS)
        if (usePre && (l + 1 < LATENT)) wait_async0();
#endif
        __syncthreads();
    }

    if (q < 8) {
        int m = wid * 16 + q + 8 * ksel;
        log_out[n0 + m] = logacc;
    }
    // coalesced residual write-back
    for (int idx = tid; idx < NTILE * LATENT; idx += NTHREADS) {
        int m = idx >> 6, l = idx & 63;
        res_out[(n0 + m) * 64 + l] = res_lds[m * 65 + l];
    }
}

// ---------------------------------------------------------------------------
extern "C" void kernel_launch(void* const* d_in, const int* in_sizes, int n_in,
                              void* d_out, int out_size, void* d_ws, size_t ws_size,
                              hipStream_t stream) {
    const float* x  = (const float*)d_in[0];
    const float* W1 = (const float*)d_in[1];
    const float* b1 = (const float*)d_in[2];
    const float* W2 = (const float*)d_in[3];
    const float* b2 = (const float*)d_in[4];

    float* res_out = (float*)d_out;
    float* log_out = res_out + (size_t)NB * LEN * LATENT;

    const size_t preBytes = (size_t)LATENT * LBLK_B;   // 2,228,224
    const bool usePre = (ws_size >= preBytes) && (d_ws != nullptr);

    if (usePre) {
        w1_split_kernel<<<dim3((SPLIT_ELEMS + 255) / 256), dim3(256), 0, stream>>>(
            W1, (unsigned short*)d_ws);
    }
    ar_flow_kernel<<<dim3(NWGS), dim3(NTHREADS), SMEM_BYTES, stream>>>(
        x, W1, b1, W2, b2, res_out, log_out,
        usePre ? (const char*)d_ws : nullptr);
}
